// OffsetGuidedSparseAttention_4045859192837
// MI455X (gfx1250) — matrix operand order repaired and checked by reading, not compiled, not run on hardware
//
#include <hip/hip_runtime.h>

// ---------------------------------------------------------------------------
// Offset-guided sparse attention for MI455X (gfx1250, wave32, WMMA).
// Pipeline:
//   1) f32->f16 converts (x, wq, wk, wv, wo, pw_w)
//   2) WMMA GEMM (f16 in, f32 acc): q_lin/k_lin/v_lin = x @ w.T + b
//   3) depthwise conv k=3 + exact GELU -> g (stored f16)
//   4) WMMA GEMM: off_lin = g @ pw_w.T + pw_b   (N = 192)
//   5) attention: tanh offsets, fractional gather+lerp of K/V, softmax over
//      R=12, weighted V sum -> attn (stored f16)
//   6) WMMA GEMM: out = attn @ wo.T + bo -> d_out (f32)
// ---------------------------------------------------------------------------

typedef _Float16 half_t;
typedef __attribute__((ext_vector_type(2)))  _Float16 v2h;
typedef __attribute__((ext_vector_type(8)))  _Float16 v8h;
typedef __attribute__((ext_vector_type(16))) _Float16 v16h;
typedef __attribute__((ext_vector_type(8)))  float    v8f;

union H16frag { v16h v; v8h h[2]; };

#define BB   2
#define QQ   2048
#define DD   1024
#define HH   16
#define RR   12
#define DH   64
#define MROW (BB * QQ)      // 4096
#define HR   (HH * RR)      // 192

// ---------------------------------------------------------------- converts
__global__ void cvt_f32_to_f16(const float* __restrict__ s,
                               half_t* __restrict__ d, int n) {
  int stride = gridDim.x * blockDim.x;
  for (int i = blockIdx.x * blockDim.x + threadIdx.x; i < n; i += stride)
    d[i] = (half_t)s[i];
}

// ------------------------------------------------------------- WMMA GEMM
// C[M,N] = A[M,K] @ W[N,K]^T + bias[N]
// A,W row-major f16 (K contiguous), C row-major f32. M%64==0, N%64==0, K%32==0.
// One wave32 computes a 64x64 tile of C (4x4 WMMA subtiles).
__global__ void __launch_bounds__(256)
wmma_gemm_f16f32(const half_t* __restrict__ A, const half_t* __restrict__ W,
                 const float* __restrict__ bias, float* __restrict__ C,
                 int M, int N, int K) {
  const int lane    = threadIdx.x & 31;
  const int wave    = blockIdx.x * (blockDim.x >> 5) + (threadIdx.x >> 5);
  const int tilesN  = N >> 6;
  const int tM      = wave / tilesN;
  const int tN      = wave - tM * tilesN;
  if (tM * 64 >= M) return;

  const int m0      = tM * 64;
  const int n0      = tN * 64;
  const int halfSel = lane >> 4;        // 0: lanes 0-15, 1: lanes 16-31
  const int lrow    = lane & 15;
  const int kA0     = halfSel * 16;     // A-matrix K phase (contiguous, matches B)
  const int kB0     = halfSel * 16;     // B-matrix K phase (contiguous, matches A)

  v8f acc[4][4];
  const v8f vzero = {0.f, 0.f, 0.f, 0.f, 0.f, 0.f, 0.f, 0.f};
#pragma unroll
  for (int mi = 0; mi < 4; ++mi)
#pragma unroll
    for (int ni = 0; ni < 4; ++ni) acc[mi][ni] = vzero;

  const half_t* Abase[4];
  const half_t* Wbase[4];
#pragma unroll
  for (int i = 0; i < 4; ++i) {
    Abase[i] = A + (size_t)(m0 + i * 16 + lrow) * K;
    Wbase[i] = W + (size_t)(n0 + i * 16 + lrow) * K;
  }

  for (int k = 0; k < K; k += 32) {
    H16frag a[4];
#pragma unroll
    for (int mi = 0; mi < 4; ++mi) {
      const half_t* p = Abase[mi] + k + kA0;
      a[mi].h[0] = *(const v8h*)(p);        // K = kA0 .. kA0+7
      a[mi].h[1] = *(const v8h*)(p + 8);    // K = kA0+8 .. kA0+15
    }
#pragma unroll
    for (int ni = 0; ni < 4; ++ni) {
      H16frag bf;
      bf.v = *(const v16h*)(Wbase[ni] + k + kB0);  // 16 contiguous halfs
#pragma unroll
      for (int mi = 0; mi < 4; ++mi) {
        acc[mi][ni] = __builtin_amdgcn_wmma_f32_16x16x32_f16(
            false, a[mi].v, false, bf.v, (short)0, acc[mi][ni], false, false);
      }
    }
  }

  // D layout: VGPR j -> row m0+mi*16 + 8*halfSel + j, col n0+ni*16 + lrow
#pragma unroll
  for (int ni = 0; ni < 4; ++ni) {
    const int col = n0 + ni * 16 + lrow;
    const float bv = bias ? bias[col] : 0.0f;
#pragma unroll
    for (int mi = 0; mi < 4; ++mi) {
      float* cp = C + (size_t)(m0 + mi * 16 + halfSel * 8) * N + col;
#pragma unroll
      for (int j = 0; j < 8; ++j) cp[(size_t)j * N] = acc[mi][ni][j] + bv;
    }
  }
}

// ------------------------------------------- depthwise conv(k=3) + GELU
// in:  q_lin (B,Q,D) f32  ->  out: g (B,Q,D) f16 (gelu(dwconv+bias))
__global__ void dwconv_gelu(const float* __restrict__ qlin,
                            const float* __restrict__ dw_w,
                            const float* __restrict__ dw_b,
                            half_t* __restrict__ g16) {
  const int total  = MROW * DD;
  const int stride = gridDim.x * blockDim.x;
  for (int i = blockIdx.x * blockDim.x + threadIdx.x; i < total; i += stride) {
    const int d  = i & (DD - 1);
    const int bq = i >> 10;            // b*Q + q
    const int qi = bq & (QQ - 1);
    const float w0 = dw_w[d * 3 + 0];
    const float w1 = dw_w[d * 3 + 1];
    const float w2 = dw_w[d * 3 + 2];
    const float xm = (qi > 0)      ? qlin[i - DD] : 0.0f;
    const float xc = qlin[i];
    const float xp = (qi < QQ - 1) ? qlin[i + DD] : 0.0f;
    float v = xm * w0 + xc * w1 + xp * w2 + dw_b[d];
    // exact GELU: 0.5*x*(1+erf(x/sqrt(2)))
    v = 0.5f * v * (1.0f + erff(v * 0.70710678118654752f));
    g16[i] = (half_t)v;
  }
}

// ------------------------------------------------------------- attention
// One wave32 per (b,h,q). Each lane owns 2 of the 64 head channels.
__global__ void __launch_bounds__(256)
offset_attention(const float* __restrict__ qlin, const float* __restrict__ klin,
                 const float* __restrict__ vlin, const float* __restrict__ off_lin,
                 const float* __restrict__ rel_scale_p,
                 half_t* __restrict__ attn16) {
  const int wid  = blockIdx.x * (blockDim.x >> 5) + (threadIdx.x >> 5);
  if (wid >= BB * HH * QQ) return;
  const int lane = threadIdx.x & 31;

  const int q = wid & (QQ - 1);
  const int h = (wid >> 11) & (HH - 1);
  const int b = wid >> 15;

  const float rs = rel_scale_p[0];
  const int c0 = h * DH + lane * 2;                   // channel pair
  const size_t rowq = ((size_t)(b * QQ + q)) * DD;

  const float2 qv = *(const float2*)(qlin + rowq + c0);

  float scores[RR], svx[RR], svy[RR];

#pragma unroll
  for (int r = 0; r < RR; ++r) {
    const float offv = off_lin[(size_t)(b * QQ + q) * HR + h * RR + r];
    const float anchor = -2.0f + (4.0f / 11.0f) * (float)r;
    float pos = (float)q + anchor + tanhf(offv) * 6.0f;
    pos = fminf(fmaxf(pos, 0.0f), (float)(QQ - 1));
    const int   lo = (int)floorf(pos);
    const int   hi = (int)ceilf(pos);
    const float fr = pos - (float)lo;

    const size_t rlo = ((size_t)(b * QQ + lo)) * DD + c0;
    const size_t rhi = ((size_t)(b * QQ + hi)) * DD + c0;

    const float2 kl = *(const float2*)(klin + rlo);
    const float2 kh = *(const float2*)(klin + rhi);
    const float sk0 = kl.x + (kh.x - kl.x) * fr;
    const float sk1 = kl.y + (kh.y - kl.y) * fr;

    const float2 vl = *(const float2*)(vlin + rlo);
    const float2 vh = *(const float2*)(vlin + rhi);
    svx[r] = vl.x + (vh.x - vl.x) * fr;
    svy[r] = vl.y + (vh.y - vl.y) * fr;

    float part = qv.x * sk0 + qv.y * sk1;
#pragma unroll
    for (int s = 16; s >= 1; s >>= 1) part += __shfl_xor(part, s, 32);
    scores[r] = part * 0.125f - rs * fabsf(pos - (float)q);
  }

  // softmax over R (uniform across lanes)
  float mx = scores[0];
#pragma unroll
  for (int r = 1; r < RR; ++r) mx = fmaxf(mx, scores[r]);
  float denom = 0.0f;
  float e[RR];
#pragma unroll
  for (int r = 0; r < RR; ++r) { e[r] = expf(scores[r] - mx); denom += e[r]; }
  const float inv = 1.0f / denom;

  float o0 = 0.0f, o1 = 0.0f;
#pragma unroll
  for (int r = 0; r < RR; ++r) {
    const float a = e[r] * inv;
    o0 += a * svx[r];
    o1 += a * svy[r];
  }

  v2h pk = {(half_t)o0, (half_t)o1};
  *(v2h*)(attn16 + rowq + c0) = pk;
}

// ---------------------------------------------------------------------------
static inline size_t alignup(size_t v, size_t a) { return (v + a - 1) & ~(a - 1); }

extern "C" void kernel_launch(void* const* d_in, const int* in_sizes, int n_in,
                              void* d_out, int out_size, void* d_ws, size_t ws_size,
                              hipStream_t stream) {
  (void)in_sizes; (void)n_in; (void)out_size; (void)ws_size;

  const float* x    = (const float*)d_in[0];
  const float* wq   = (const float*)d_in[1];
  const float* bq   = (const float*)d_in[2];
  const float* wk   = (const float*)d_in[3];
  const float* bk   = (const float*)d_in[4];
  const float* wv   = (const float*)d_in[5];
  const float* bv   = (const float*)d_in[6];
  const float* wo   = (const float*)d_in[7];
  const float* bo   = (const float*)d_in[8];
  const float* dw_w = (const float*)d_in[9];
  const float* dw_b = (const float*)d_in[10];
  const float* pw_w = (const float*)d_in[11];
  const float* pw_b = (const float*)d_in[12];
  const float* rsc  = (const float*)d_in[13];
  float* out = (float*)d_out;

  // workspace carve-up
  char* p = (char*)d_ws;
  auto take = [&](size_t bytes) { char* r = p; p += alignup(bytes, 256); return r; };
  half_t* x16    = (half_t*)take((size_t)MROW * DD * 2);
  half_t* wq16   = (half_t*)take((size_t)DD * DD * 2);
  half_t* wk16   = (half_t*)take((size_t)DD * DD * 2);
  half_t* wv16   = (half_t*)take((size_t)DD * DD * 2);
  half_t* wo16   = (half_t*)take((size_t)DD * DD * 2);
  half_t* pw16   = (half_t*)take((size_t)HR * DD * 2);
  float*  qlin   = (float*)take((size_t)MROW * DD * 4);
  float*  klin   = (float*)take((size_t)MROW * DD * 4);
  float*  vlin   = (float*)take((size_t)MROW * DD * 4);
  half_t* g16    = (half_t*)take((size_t)MROW * DD * 2);
  float*  offl   = (float*)take((size_t)MROW * HR * 4);
  half_t* attn16 = (half_t*)take((size_t)MROW * DD * 2);

  // 1) conversions
  {
    int n = MROW * DD;
    cvt_f32_to_f16<<<(n + 255) / 256, 256, 0, stream>>>(x, x16, n);
    n = DD * DD;
    cvt_f32_to_f16<<<(n + 255) / 256, 256, 0, stream>>>(wq, wq16, n);
    cvt_f32_to_f16<<<(n + 255) / 256, 256, 0, stream>>>(wk, wk16, n);
    cvt_f32_to_f16<<<(n + 255) / 256, 256, 0, stream>>>(wv, wv16, n);
    cvt_f32_to_f16<<<(n + 255) / 256, 256, 0, stream>>>(wo, wo16, n);
    n = HR * DD;
    cvt_f32_to_f16<<<(n + 255) / 256, 256, 0, stream>>>(pw_w, pw16, n);
  }

  // 2) QKV projections: (4096 x 1024) = x16 @ w.T + b
  {
    const int waves = (MROW / 64) * (DD / 64);     // 1024
    const int blocks = (waves + 7) / 8;            // 8 waves / block
    wmma_gemm_f16f32<<<blocks, 256, 0, stream>>>(x16, wq16, bq, qlin, MROW, DD, DD);
    wmma_gemm_f16f32<<<blocks, 256, 0, stream>>>(x16, wk16, bk, klin, MROW, DD, DD);
    wmma_gemm_f16f32<<<blocks, 256, 0, stream>>>(x16, wv16, bv, vlin, MROW, DD, DD);
  }

  // 3) depthwise conv + gelu -> g16
  {
    int n = MROW * DD;
    dwconv_gelu<<<(n + 255) / 256, 256, 0, stream>>>(qlin, dw_w, dw_b, g16);
  }

  // 4) pointwise projection to offsets: (4096 x 192) = g16 @ pw_w.T + pw_b
  {
    const int waves = (MROW / 64) * (HR / 64);     // 192
    const int blocks = (waves + 7) / 8;
    wmma_gemm_f16f32<<<blocks, 256, 0, stream>>>(g16, pw16, pw_b, offl, MROW, HR, DD);
  }

  // 5) offset-guided attention -> attn16
  {
    const int waves = BB * HH * QQ;                // 65536
    const int blocks = waves / 8;
    offset_attention<<<blocks, 256, 0, stream>>>(qlin, klin, vlin, offl, rsc, attn16);
  }

  // 6) output projection -> d_out
  {
    const int waves = (MROW / 64) * (DD / 64);
    const int blocks = (waves + 7) / 8;
    wmma_gemm_f16f32<<<blocks, 256, 0, stream>>>(attn16, wo16, bo, out, MROW, DD, DD);
  }
}
